// PyGRGCNModel_55817394978950
// MI455X (gfx1250) — matrix operand order repaired and checked by reading, not compiled, not run on hardware
//
#include <hip/hip_runtime.h>
#include <hip/hip_bf16.h>
#include <stdint.h>

typedef __bf16 bf16_t;
typedef __attribute__((ext_vector_type(16))) __bf16 v16bf;
typedef __attribute__((ext_vector_type(8)))  __bf16 v8bf;
typedef __attribute__((ext_vector_type(4)))  __bf16 v4bf;
typedef __attribute__((ext_vector_type(8)))  float  v8f;
typedef __attribute__((ext_vector_type(4)))  unsigned tdm_u4;
typedef __attribute__((ext_vector_type(8)))  int      tdm_i8;
typedef __attribute__((ext_vector_type(4)))  int      tdm_i4;

#define NNODES 100000
#define NEDGES 1600000
#define NREL   3

#if defined(__has_builtin)
#if __has_builtin(__builtin_amdgcn_tensor_load_to_lds) && __has_builtin(__builtin_amdgcn_s_wait_tensorcnt)
#define USE_TDM 1
#endif
#endif

// ---------------- Tensor Data Mover: 2D tile (tile == tensor window) -> LDS ----------------
#if defined(USE_TDM)
__device__ __forceinline__ void tdm_load_2d(unsigned lds_off, const void* gptr,
                                            unsigned esz_code,  // 0=1B 1=2B 2=4B 3=8B
                                            unsigned dim0, unsigned dim1, unsigned stride0) {
  uint64_t ga = (uint64_t)(uintptr_t)gptr;
  tdm_u4 g0;
  g0[0] = 1u;                                                   // count=1, user descriptor
  g0[1] = lds_off;                                              // LDS byte address
  g0[2] = (unsigned)ga;                                         // global_addr[31:0]
  g0[3] = (unsigned)((ga >> 32) & 0x01FFFFFFu) | 0x80000000u;   // global_addr[56:32] | type=2
  tdm_i8 g1;
  g1[0] = (int)(esz_code << 16);                                // mask=0, data_size
  g1[1] = (int)(dim0 << 16);                                    // tensor_dim0[15:0]
  g1[2] = (int)((dim0 >> 16) | (dim1 << 16));                   // tensor_dim0[31:16] | tensor_dim1[15:0]
  g1[3] = (int)(((dim1 >> 16) & 0xFFFFu) | (dim0 << 16));       // tensor_dim1[31:16] | tile_dim0
  g1[4] = (int)(dim1 & 0xFFFFu);                                // tile_dim1 | tile_dim2=0
  g1[5] = (int)stride0;                                         // tensor_dim0_stride[31:0]
  g1[6] = 0;                                                    // stride0 hi | dim1_stride lo
  g1[7] = 0;
  tdm_i4 gz = {0, 0, 0, 0};
#if __clang_major__ >= 23
  tdm_i8 gz8 = {0, 0, 0, 0, 0, 0, 0, 0};
  __builtin_amdgcn_tensor_load_to_lds(g0, g1, gz, gz, gz8, 0);
#else
  __builtin_amdgcn_tensor_load_to_lds(g0, g1, gz, gz, 0);
#endif
}
#endif

// ---------------- elementwise helpers ----------------

__global__ void k_fill(float* __restrict__ p, int n, float v) {
  int i = blockIdx.x * blockDim.x + threadIdx.x;
  if (i < n) p[i] = v;
}

__global__ void k_deg_scatter(const int* __restrict__ dst, float* __restrict__ deg, int nE) {
  int e = blockIdx.x * blockDim.x + threadIdx.x;
  if (e < nE) unsafeAtomicAdd(&deg[dst[e]], 1.0f);
}

__global__ void k_rsqrt(float* __restrict__ p, int n) {
  int i = blockIdx.x * blockDim.x + threadIdx.x;
  if (i < n) p[i] = __frsqrt_rn(p[i]);
}

__global__ void k_selfloop_bias(const float* __restrict__ ht, const float* __restrict__ dinv,
                                const float* __restrict__ bias, float* __restrict__ acc,
                                int total, int F) {
  int i = blockIdx.x * blockDim.x + threadIdx.x;
  if (i >= total) return;
  int node = i / F;
  int f = i - node * F;
  float s = dinv[node];
  acc[i] += ht[i] * (s * s) + bias[f];
}

__global__ void k_finalize(float* __restrict__ buf, int n, float scale, int relu) {
  int i = blockIdx.x * blockDim.x + threadIdx.x;
  if (i >= n) return;
  float v = buf[i] * scale;
  buf[i] = relu ? fmaxf(v, 0.0f) : v;
}

// ---------------- pack W (f32, [128 x NOUT]) into bf16 WMMA B-fragment order ----------------
// Wpk[((nt*4 + kbi)*32 + lane)*16 + i] = bf16( W[kbi*32 + 16*(lane>>4) + i][nt*16 + (lane&15)] )

__global__ void k_pack_w(const float* __restrict__ W, bf16_t* __restrict__ Wpk, int nout) {
  int idx = blockIdx.x * blockDim.x + threadIdx.x;
  if (idx >= nout * 128) return;
  int i    = idx & 15;
  int lane = (idx >> 4) & 31;
  int kbi  = (idx >> 9) & 3;
  int nt   = idx >> 11;
  int k    = kbi * 32 + (lane >> 4) * 16 + i;
  int n    = nt * 16 + (lane & 15);
  Wpk[idx] = (bf16_t)W[(size_t)k * nout + n];
}

// ---------------- WMMA GEMM: H[N x NOUT] = X[N x 128] * W[128 x NOUT] ----------------
// Block = 16 output rows, (NOUT/16) waves (one 16x16 tile each).
// TDM stages the 16x128 f32 X tile and the packed bf16 W into LDS; X converted to bf16
// once per block; inner loop is pure ds_load_b128 + v_wmma_f32_16x16x32_bf16.

template<int NOUT>
__launch_bounds__(32 * (NOUT / 16))
__global__ void k_gemm_wmma(const float* __restrict__ X, const bf16_t* __restrict__ Wpk,
                            float* __restrict__ H) {
  __shared__ float  xtile[16 * 128];            // 8 KB
  __shared__ bf16_t xbf[16 * 128];              // 4 KB
  __shared__ bf16_t wbf[NOUT * 128];            // 32 KB / 16 KB

  const int tid  = threadIdx.x;
  const int lane = tid & 31;
  const int wave = tid >> 5;
  const int m0   = blockIdx.x << 4;             // 100000 = 6250*16: always full tiles

#if defined(USE_TDM)
  if (tid < 32) {  // one wave issues both DMAs (TDM ignores EXEC; gate per-wave)
    tdm_load_2d((unsigned)(uintptr_t)xtile, X + (size_t)m0 * 128,
                /*f32*/2, /*dim0*/128, /*dim1*/16, /*stride0*/128);
    tdm_load_2d((unsigned)(uintptr_t)wbf, Wpk,
                /*bf16*/1, /*dim0*/2048, /*dim1*/NOUT / 16, /*stride0*/2048);
    __builtin_amdgcn_s_wait_tensorcnt(0);
  }
  __syncthreads();
#else
  for (int i = tid; i < 16 * 128 / 4; i += blockDim.x)
    reinterpret_cast<float4*>(xtile)[i] =
        reinterpret_cast<const float4*>(X + (size_t)m0 * 128)[i];
  for (int i = tid; i < NOUT * 128 / 8; i += blockDim.x)
    reinterpret_cast<v8bf*>(wbf)[i] = reinterpret_cast<const v8bf*>(Wpk)[i];
  __syncthreads();
#endif

  // f32 -> bf16 conversion of the X tile, once per block (v_cvt_pk_bf16_f32 + ds_store_b64)
  for (int i = tid; i < 16 * 128 / 4; i += blockDim.x) {
    const float4 f = reinterpret_cast<const float4*>(xtile)[i];
    v4bf p;
    p[0] = (bf16_t)f.x; p[1] = (bf16_t)f.y; p[2] = (bf16_t)f.z; p[3] = (bf16_t)f.w;
    reinterpret_cast<v4bf*>(xbf)[i] = p;
  }
  __syncthreads();

  const int n0   = wave << 4;
  const int lrow = lane & 15;
  const int hi   = lane >> 4;
  const bf16_t* __restrict__ arow = xbf + lrow * 128;
  const bf16_t* __restrict__ bp   = wbf + (size_t)(n0 >> 4) * 2048 + lane * 16;

  v8f acc = {};
#pragma unroll
  for (int kbi = 0; kbi < 4; ++kbi) {
    const int kb = kbi * 32;
    // A fragment (16x32 bf16): lane-half hi holds K = kb+16*hi+{0..15}
    const v8bf a0 = *reinterpret_cast<const v8bf*>(arow + kb + hi * 16);
    const v8bf a1 = *reinterpret_cast<const v8bf*>(arow + kb + hi * 16 + 8);
    // B fragment (32x16 bf16): lane = column, lane-half = contiguous K range of 16 (pre-packed)
    const v16bf b = *reinterpret_cast<const v16bf*>(bp + (size_t)kbi * 512);
    v16bf a;
#pragma unroll
    for (int i = 0; i < 8; ++i) { a[i] = a0[i]; a[8 + i] = a1[i]; }
    acc = __builtin_amdgcn_wmma_f32_16x16x32_bf16(
        false, a, false, b, (short)0, acc, false, false);
  }
  // C layout: VGPR v -> row m0 + v + 8*hi, col n0 + (lane&15)
  float* __restrict__ hp = H + (size_t)(m0 + hi * 8) * NOUT + n0 + lrow;
#pragma unroll
  for (int v = 0; v < 8; ++v) hp[(size_t)v * NOUT] = acc[v];
}

// ---------------- edge scatter: acc[dst] += ht[src] * dinv[src]*dinv[dst] ----------------

template<int CSHIFT>
__global__ void k_edge_scatter(const float* __restrict__ ht, const int* __restrict__ src,
                               const int* __restrict__ dst, const float* __restrict__ dinv,
                               float* __restrict__ acc, int nE) {
  int tid = blockIdx.x * blockDim.x + threadIdx.x;
  int e = tid >> CSHIFT;
  int c = tid & ((1 << CSHIFT) - 1);
  if (e >= nE) return;
  if (c == 0) {
    __builtin_prefetch(src + e + 4096, 0, 0);
    __builtin_prefetch(dst + e + 4096, 0, 0);
  }
  int s = src[e];
  int d = dst[e];
  float coef = dinv[s] * dinv[d];
  const int F = 4 << CSHIFT;
  const float4 v = *reinterpret_cast<const float4*>(ht + (size_t)s * F + (c << 2));
  float* p = acc + (size_t)d * F + (c << 2);
  unsafeAtomicAdd(p + 0, v.x * coef);
  unsafeAtomicAdd(p + 1, v.y * coef);
  unsafeAtomicAdd(p + 2, v.z * coef);
  unsafeAtomicAdd(p + 3, v.w * coef);
}

// ---------------- driver ----------------

extern "C" void kernel_launch(void* const* d_in, const int* in_sizes, int n_in,
                              void* d_out, int out_size, void* d_ws, size_t ws_size,
                              hipStream_t stream) {
  const float* x     = (const float*)d_in[0];   // [N,128]
  const int*   edges = (const int*)  d_in[1];   // [T,2,E]
  const float* W1    = (const float*)d_in[2];   // [T,128,128]
  const float* b1    = (const float*)d_in[3];   // [T,128]
  const float* W2    = (const float*)d_in[4];   // [T,128,64]
  const float* b2    = (const float*)d_in[5];   // [T,64]
  float* out = (float*)d_out;                   // [N,64]

  float*  ws   = (float*)d_ws;
  float*  dinv = ws;                                   // NREL*NNODES
  float*  ht   = dinv + (size_t)NREL * NNODES;         // NNODES*128
  float*  acc  = ht   + (size_t)NNODES * 128;          // NNODES*128
  bf16_t* wpk  = (bf16_t*)(acc + (size_t)NNODES * 128);// up to 128*128 bf16

  const int B = 256;
  auto cdiv = [](long a, long b) { return (int)((a + b - 1) / b); };

  // --- symmetric-norm degree factors (same edges for both layers) ---
  k_fill<<<cdiv((long)NREL * NNODES, B), B, 0, stream>>>(dinv, NREL * NNODES, 1.0f);
  for (int t = 0; t < NREL; ++t) {
    const int* dst = edges + (size_t)t * 2 * NEDGES + NEDGES;
    k_deg_scatter<<<cdiv(NEDGES, B), B, 0, stream>>>(dst, dinv + (size_t)t * NNODES, NEDGES);
  }
  k_rsqrt<<<cdiv((long)NREL * NNODES, B), B, 0, stream>>>(dinv, NREL * NNODES);

  // --- layer 1: 128 -> 128, mean over relations, ReLU ---
  k_fill<<<cdiv((long)NNODES * 128, B), B, 0, stream>>>(acc, NNODES * 128, 0.0f);
  for (int t = 0; t < NREL; ++t) {
    const int* src = edges + (size_t)t * 2 * NEDGES;
    const int* dst = src + NEDGES;
    const float* dv = dinv + (size_t)t * NNODES;
    k_pack_w<<<cdiv(128 * 128, B), B, 0, stream>>>(W1 + (size_t)t * 128 * 128, wpk, 128);
    k_gemm_wmma<128><<<NNODES / 16, 256, 0, stream>>>(x, wpk, ht);
    k_selfloop_bias<<<cdiv((long)NNODES * 128, B), B, 0, stream>>>(
        ht, dv, b1 + (size_t)t * 128, acc, NNODES * 128, 128);
    k_edge_scatter<5><<<cdiv((long)NEDGES << 5, B), B, 0, stream>>>(ht, src, dst, dv, acc, NEDGES);
  }
  k_finalize<<<cdiv((long)NNODES * 128, B), B, 0, stream>>>(acc, NNODES * 128, 1.0f / 3.0f, 1);

  // --- layer 2: 128 -> 64, mean over relations, no ReLU; accumulate into d_out ---
  k_fill<<<cdiv((long)NNODES * 64, B), B, 0, stream>>>(out, NNODES * 64, 0.0f);
  for (int t = 0; t < NREL; ++t) {
    const int* src = edges + (size_t)t * 2 * NEDGES;
    const int* dst = src + NEDGES;
    const float* dv = dinv + (size_t)t * NNODES;
    k_pack_w<<<cdiv(64 * 128, B), B, 0, stream>>>(W2 + (size_t)t * 128 * 64, wpk, 64);
    k_gemm_wmma<64><<<NNODES / 16, 128, 0, stream>>>(acc, wpk, ht);
    k_selfloop_bias<<<cdiv((long)NNODES * 64, B), B, 0, stream>>>(
        ht, dv, b2 + (size_t)t * 64, out, NNODES * 64, 64);
    k_edge_scatter<4><<<cdiv((long)NEDGES << 4, B), B, 0, stream>>>(ht, src, dst, dv, out, NEDGES);
  }
  k_finalize<<<cdiv((long)NNODES * 64, B), B, 0, stream>>>(out, NNODES * 64, 1.0f / 3.0f, 0);
}
